// SelfAttention_56564719289016
// MI455X (gfx1250) — matrix operand order repaired and checked by reading, not compiled, not run on hardware
//
#include <hip/hip_runtime.h>

// ---------------------------------------------------------------------------
// Self-attention (B=8, C=128, N=4096, DQK=32) for MI455X / gfx1250 (wave32).
// bf16 WMMA 16x16x32 for energy (Q·Q^T), V·P, and the trans-conv GEMM.
// Two-pass flash-style softmax: pass1 row stats, pass2 recompute + accumulate.
// Software-pipelined so independent WMMAs cover the WMMA->VALU hazard window.
// ---------------------------------------------------------------------------

#define BATCH 8
#define CH    128
#define NN    4096
#define DQK   32

__constant__ const float kScale = 0.17677669529663687f;  // 1/sqrt(32)

typedef __bf16 v8bf  __attribute__((ext_vector_type(8)));
typedef __bf16 v16bf __attribute__((ext_vector_type(16)));
typedef float  v8f   __attribute__((ext_vector_type(8)));

__device__ __forceinline__ __bf16 f2bf(float f) {
  unsigned u = __builtin_bit_cast(unsigned, f);
  u += 0x7FFFu + ((u >> 16) & 1u);           // round-to-nearest-even
  unsigned short h = (unsigned short)(u >> 16);
  return __builtin_bit_cast(__bf16, h);
}

// A-matrix tile (16x32 bf16, M = lane%16). Lanes 0-15 hold K 0-15,
// lanes 16-31 hold K 16-31 (contiguous order, matching the B operands).
__device__ __forceinline__ v16bf load_a_tile(const __bf16* rowp, int hi) {
  const int o = hi ? 16 : 0;
  v8bf lo = *(const v8bf*)(rowp + o);
  v8bf hh = *(const v8bf*)(rowp + 8 + o);
  return __builtin_shufflevector(lo, hh, 0, 1, 2, 3, 4, 5, 6, 7,
                                 8, 9, 10, 11, 12, 13, 14, 15);
}

__device__ __forceinline__ v8f wmma_bf16(v16bf a, v16bf b, v8f c) {
  return __builtin_amdgcn_wmma_f32_16x16x32_bf16(false, a, false, b,
                                                 (short)0, c, false, false);
}

// ---------------------------------------------------------------------------
// Kernel 1: fp32 projections -> bf16 q (B,N,32) and v (B,C,N).
// ---------------------------------------------------------------------------
__global__ __launch_bounds__(256) void proj_kernel(
    const float* __restrict__ x, const float* __restrict__ qk_w,
    const float* __restrict__ v_w, const float* __restrict__ v_b,
    __bf16* __restrict__ qb, __bf16* __restrict__ vb) {
  __shared__ float xs[CH * 64];
  const int b   = blockIdx.x / (NN / 64);
  const int n0  = (blockIdx.x % (NN / 64)) * 64;
  const int tid = threadIdx.x;
  for (int i = tid; i < CH * 64; i += 256) {
    int c = i >> 6, nl = i & 63;
    xs[i] = x[((size_t)b * CH + c) * NN + n0 + nl];
  }
  __syncthreads();
  for (int oi = tid; oi < (DQK + CH) * 64; oi += 256) {
    int o = oi >> 6, nl = oi & 63;
    float acc = 0.f;
    if (o < DQK) {
      const float* w = qk_w + o * CH;
      for (int c = 0; c < CH; ++c) acc += w[c] * xs[(c << 6) + nl];
      qb[((size_t)b * NN + n0 + nl) * DQK + o] = f2bf(acc);
    } else {
      const int co   = o - DQK;
      const float* w = v_w + co * CH;
      acc = v_b[co];
      for (int c = 0; c < CH; ++c) acc += w[c] * xs[(c << 6) + nl];
      vb[((size_t)b * CH + co) * NN + n0 + nl] = f2bf(acc);
    }
  }
}

__global__ __launch_bounds__(256) void cvt_tw_kernel(
    const float* __restrict__ t_w, __bf16* __restrict__ twb) {
  int i = blockIdx.x * 256 + threadIdx.x;
  if (i < CH * CH) twb[i] = f2bf(t_w[i]);
}

// ---------------------------------------------------------------------------
// Kernel 2: softmax row stats. One wave owns 32 query rows; streams all 4096
// keys through 16-wide B tiles. Pipelined: WMMAs for tile i+1 issue before the
// online max/sum update of tile i, hiding the WMMA->VALU hazard.
// ---------------------------------------------------------------------------
__global__ __launch_bounds__(256) void softmax_stats_kernel(
    const __bf16* __restrict__ qb, float* __restrict__ rmax,
    float* __restrict__ rinv) {
  const int wave = blockIdx.x * 8 + (threadIdx.x >> 5);
  const int lane = threadIdx.x & 31;
  const int b    = wave / (NN / 32);
  const int n0   = (wave % (NN / 32)) * 32;
  const int hi   = lane >> 4;
  const int col  = lane & 15;
  const __bf16* qbase = qb + (size_t)b * NN * DQK;

  v16bf a0 = load_a_tile(qbase + (size_t)(n0 + col) * DQK, hi);
  v16bf a1 = load_a_tile(qbase + (size_t)(n0 + 16 + col) * DQK, hi);

  float m0[8], s0[8], m1[8], s1[8];
  for (int r = 0; r < 8; ++r) { m0[r] = m1[r] = -3.0e38f; s0[r] = s1[r] = 0.f; }

  v8f z = {};
  v16bf bt = *(const v16bf*)(qbase + (size_t)col * DQK + hi * 16);
  v8f e0 = wmma_bf16(a0, bt, z);
  v8f e1 = wmma_bf16(a1, bt, z);
  for (int mm = 16; mm <= NN; mm += 16) {
    v8f f0 = z, f1 = z;
    if (mm < NN) {  // issue next tile's WMMAs before consuming e0/e1
      v16bf btn = *(const v16bf*)(qbase + (size_t)(mm + col) * DQK + hi * 16);
      f0 = wmma_bf16(a0, btn, z);
      f1 = wmma_bf16(a1, btn, z);
    }
    for (int r = 0; r < 8; ++r) {
      float e  = e0[r] * kScale;
      float nm = fmaxf(m0[r], e);
      s0[r] = s0[r] * __expf(m0[r] - nm) + __expf(e - nm);
      m0[r] = nm;
      e  = e1[r] * kScale;
      nm = fmaxf(m1[r], e);
      s1[r] = s1[r] * __expf(m1[r] - nm) + __expf(e - nm);
      m1[r] = nm;
    }
    e0 = f0;
    e1 = f1;
  }
  // combine partials across the 16 lanes that share the same rows
  for (int mask = 1; mask < 16; mask <<= 1) {
    for (int r = 0; r < 8; ++r) {
      float om = __shfl_xor(m0[r], mask, 32);
      float os = __shfl_xor(s0[r], mask, 32);
      float nm = fmaxf(m0[r], om);
      s0[r] = s0[r] * __expf(m0[r] - nm) + os * __expf(om - nm);
      m0[r] = nm;
      om = __shfl_xor(m1[r], mask, 32);
      os = __shfl_xor(s1[r], mask, 32);
      nm = fmaxf(m1[r], om);
      s1[r] = s1[r] * __expf(m1[r] - nm) + os * __expf(om - nm);
      m1[r] = nm;
    }
  }
  if (col == 0) {
    for (int r = 0; r < 8; ++r) {
      int row0 = n0 + hi * 8 + r;
      rmax[(size_t)b * NN + row0] = m0[r];
      rinv[(size_t)b * NN + row0] = 1.0f / s0[r];
      int row1 = n0 + 16 + hi * 8 + r;
      rmax[(size_t)b * NN + row1] = m1[r];
      rinv[(size_t)b * NN + row1] = 1.0f / s1[r];
    }
  }
}

// ---------------------------------------------------------------------------
// Kernel 3: x_s[:, m] = sum_n attn[n,m] * v[:, n]. One wave owns 32 output
// columns (two 16-wide m tiles) x all 128 channels (16 accumulator tiles).
// Per 32-key step: 4 energy WMMAs (pipelined one step ahead), weight tiles via
// half-wave shuffles, 16 PV WMMAs reusing each V A-tile twice.
// Output stored transposed (B,N,C) in bf16.
// ---------------------------------------------------------------------------
__global__ __launch_bounds__(256) void attn_pv_kernel(
    const __bf16* __restrict__ qb, const __bf16* __restrict__ vb,
    const float* __restrict__ rmax, const float* __restrict__ rinv,
    __bf16* __restrict__ xs) {
  const int wave = blockIdx.x * 8 + (threadIdx.x >> 5);
  const int lane = threadIdx.x & 31;
  const int b    = wave / (NN / 32);
  const int m0   = (wave % (NN / 32)) * 32;
  const int hi   = lane >> 4;
  const int col  = lane & 15;
  const __bf16* qbase = qb + (size_t)b * NN * DQK;
  const __bf16* vbase = vb + (size_t)b * CH * NN;
  const float*  rm    = rmax + (size_t)b * NN;
  const float*  ri    = rinv + (size_t)b * NN;

  v16bf bm0 = *(const v16bf*)(qbase + (size_t)(m0 + col) * DQK + hi * 16);
  v16bf bm1 = *(const v16bf*)(qbase + (size_t)(m0 + 16 + col) * DQK + hi * 16);
  v8f z = {};
  v8f acc0[8], acc1[8];
  for (int t = 0; t < 8; ++t) { acc0[t] = z; acc1[t] = z; }

  // prologue: energy tiles for n0 = 0
  v16bf a0 = load_a_tile(qbase + (size_t)col * DQK, hi);
  v16bf a1 = load_a_tile(qbase + (size_t)(16 + col) * DQK, hi);
  v8f e00 = wmma_bf16(a0, bm0, z);   // rows n0..n0+15, cols m0 tile
  v8f e10 = wmma_bf16(a1, bm0, z);   // rows n0+16..n0+31
  v8f e01 = wmma_bf16(a0, bm1, z);   // cols m0+16 tile
  v8f e11 = wmma_bf16(a1, bm1, z);

  for (int n0 = 0; n0 < NN; n0 += 32) {
    // softmax weights for current 32 rows x 32 cols
    float w00[8], w10[8], w01[8], w11[8];
    for (int r = 0; r < 8; ++r) {
      int r0 = n0 + hi * 8 + r;
      int r1 = r0 + 16;
      float mx0 = rm[r0], iv0 = ri[r0];
      float mx1 = rm[r1], iv1 = ri[r1];
      w00[r] = __expf(e00[r] * kScale - mx0) * iv0;
      w01[r] = __expf(e01[r] * kScale - mx0) * iv0;
      w10[r] = __expf(e10[r] * kScale - mx1) * iv1;
      w11[r] = __expf(e11[r] * kScale - mx1) * iv1;
    }
    // D-layout (8 rows/half-lane) -> B-layout (16 rows/lane) via xor-16 swap
    v16bf wb0, wb1;
    for (int r = 0; r < 8; ++r) {
      float o00 = __shfl_xor(w00[r], 16, 32);
      float o10 = __shfl_xor(w10[r], 16, 32);
      float o01 = __shfl_xor(w01[r], 16, 32);
      float o11 = __shfl_xor(w11[r], 16, 32);
      wb0[r]     = f2bf(hi ? o10 : w00[r]);
      wb0[8 + r] = f2bf(hi ? w10[r] : o00);
      wb1[r]     = f2bf(hi ? o11 : w01[r]);
      wb1[8 + r] = f2bf(hi ? w11[r] : o01);
    }
    // issue next step's energy WMMAs early: the 16 PV WMMAs below cover the
    // WMMA->VALU hazard before next iteration's __expf consumes them
    if (n0 + 32 < NN) {
      a0  = load_a_tile(qbase + (size_t)(n0 + 32 + col) * DQK, hi);
      a1  = load_a_tile(qbase + (size_t)(n0 + 48 + col) * DQK, hi);
      e00 = wmma_bf16(a0, bm0, z);
      e10 = wmma_bf16(a1, bm0, z);
      e01 = wmma_bf16(a0, bm1, z);
      e11 = wmma_bf16(a1, bm1, z);
    }
    for (int t = 0; t < 8; ++t) {
      v16bf av = load_a_tile(vbase + (size_t)(t * 16 + col) * NN + n0, hi);
      acc0[t] = wmma_bf16(av, wb0, acc0[t]);
      acc1[t] = wmma_bf16(av, wb1, acc1[t]);
    }
  }
  __bf16* outp0 = xs + ((size_t)b * NN + m0 + col) * CH;
  __bf16* outp1 = xs + ((size_t)b * NN + m0 + 16 + col) * CH;
  for (int t = 0; t < 8; ++t) {
    v8bf o0, o1;
    for (int r = 0; r < 8; ++r) {
      o0[r] = f2bf(acc0[t][r]);
      o1[r] = f2bf(acc1[t][r]);
    }
    *(v8bf*)(outp0 + t * 16 + hi * 8) = o0;
    *(v8bf*)(outp1 + t * 16 + hi * 8) = o1;
  }
}

// ---------------------------------------------------------------------------
// Kernel 4: y = t_w @ x_s + t_b (WMMA, K=128 in 4 steps). y stored (B,N,C).
// ---------------------------------------------------------------------------
__global__ __launch_bounds__(256) void trans_gemm_kernel(
    const __bf16* __restrict__ twb, const __bf16* __restrict__ xs,
    const float* __restrict__ t_b, float* __restrict__ yt) {
  const int wave = blockIdx.x * 8 + (threadIdx.x >> 5);
  const int lane = threadIdx.x & 31;
  const int b    = wave / (NN / 16);
  const int n0   = (wave % (NN / 16)) * 16;
  const int hi   = lane >> 4;
  const int col  = lane & 15;
  const __bf16* xrow = xs + ((size_t)b * NN + n0 + col) * CH;

  v8f acc[8];
  v8f z = {};
  for (int t = 0; t < 8; ++t) acc[t] = z;
  for (int k0 = 0; k0 < CH; k0 += 32) {
    v16bf bt = *(const v16bf*)(xrow + k0 + hi * 16);
    for (int t = 0; t < 8; ++t) {
      v16bf at = load_a_tile(twb + (size_t)(t * 16 + col) * CH + k0, hi);
      acc[t] = wmma_bf16(at, bt, acc[t]);
    }
  }
  float* outp = yt + ((size_t)b * NN + n0 + col) * CH;
  for (int t = 0; t < 8; ++t) {
    v8f o;
    for (int r = 0; r < 8; ++r) o[r] = acc[t][r] + t_b[t * 16 + hi * 8 + r];
    *(v8f*)(outp + t * 16 + hi * 8) = o;
  }
}

// ---------------------------------------------------------------------------
// Kernel 5: per-channel batch stats (mean, rstd) over (B, N).
// ---------------------------------------------------------------------------
__global__ __launch_bounds__(256) void bn_stats_kernel(
    const float* __restrict__ yt, float* __restrict__ stats) {
  const int o   = blockIdx.x;
  const int tid = threadIdx.x;
  float p1 = 0.f, p2 = 0.f;
  for (int i = tid; i < BATCH * NN; i += 256) {
    float v = yt[(size_t)i * CH + o];
    p1 += v;
    p2 += v * v;
  }
  __shared__ float a1[256], a2[256];
  a1[tid] = p1;
  a2[tid] = p2;
  __syncthreads();
  for (int st = 128; st > 0; st >>= 1) {
    if (tid < st) { a1[tid] += a1[tid + st]; a2[tid] += a2[tid + st]; }
    __syncthreads();
  }
  if (tid == 0) {
    const float inv = 1.0f / (float)(BATCH * NN);
    float mean = a1[0] * inv;
    float var  = a2[0] * inv - mean * mean;
    stats[o]      = mean;
    stats[CH + o] = rsqrtf(var + 1e-5f);
  }
}

// ---------------------------------------------------------------------------
// Kernel 6: out = x + relu(gamma * (y - mean) * rstd + beta).
// ---------------------------------------------------------------------------
__global__ __launch_bounds__(256) void finalize_kernel(
    const float* __restrict__ x, const float* __restrict__ yt,
    const float* __restrict__ stats, const float* __restrict__ gamma,
    const float* __restrict__ beta, float* __restrict__ out) {
  size_t idx = (size_t)blockIdx.x * 256 + threadIdx.x;
  if (idx >= (size_t)BATCH * CH * NN) return;
  int n = (int)(idx % NN);
  int c = (int)((idx / NN) % CH);
  int b = (int)(idx / ((size_t)NN * CH));
  float y  = yt[((size_t)b * NN + n) * CH + c];
  float yh = (y - stats[c]) * stats[CH + c];
  float v  = gamma[c] * yh + beta[c];
  v = v > 0.f ? v : 0.f;
  out[idx] = x[idx] + v;
}

// ---------------------------------------------------------------------------
extern "C" void kernel_launch(void* const* d_in, const int* in_sizes, int n_in,
                              void* d_out, int out_size, void* d_ws,
                              size_t ws_size, hipStream_t stream) {
  (void)in_sizes; (void)n_in; (void)out_size; (void)ws_size;
  const float* x     = (const float*)d_in[0];
  const float* qk_w  = (const float*)d_in[1];
  const float* v_w   = (const float*)d_in[2];
  const float* v_b   = (const float*)d_in[3];
  const float* t_w   = (const float*)d_in[4];
  const float* t_b   = (const float*)d_in[5];
  const float* gamma = (const float*)d_in[6];
  const float* beta  = (const float*)d_in[7];
  float* out = (float*)d_out;

  char* ws = (char*)d_ws;
  size_t off = 0;
  __bf16* qb  = (__bf16*)(ws + off); off += (size_t)BATCH * NN * DQK * 2;  // 2 MB
  __bf16* vb  = (__bf16*)(ws + off); off += (size_t)BATCH * CH * NN * 2;   // 8.4 MB
  __bf16* twb = (__bf16*)(ws + off); off += (size_t)CH * CH * 2;           // 32 KB
  float*  rmx = (float*)(ws + off);  off += (size_t)BATCH * NN * 4;        // 128 KB
  float*  rin = (float*)(ws + off);  off += (size_t)BATCH * NN * 4;        // 128 KB
  __bf16* xs  = (__bf16*)(ws + off); off += (size_t)BATCH * NN * CH * 2;   // 8.4 MB
  float*  yt  = (float*)(ws + off);  off += (size_t)BATCH * NN * CH * 4;   // 16.8 MB
  float*  st  = (float*)(ws + off);  off += 2 * CH * 4;                    // 1 KB

  proj_kernel<<<BATCH * (NN / 64), 256, 0, stream>>>(x, qk_w, v_w, v_b, qb, vb);
  cvt_tw_kernel<<<(CH * CH + 255) / 256, 256, 0, stream>>>(t_w, twb);
  softmax_stats_kernel<<<BATCH * (NN / 32) / 8, 256, 0, stream>>>(qb, rmx, rin);
  attn_pv_kernel<<<BATCH * (NN / 32) / 8, 256, 0, stream>>>(qb, vb, rmx, rin, xs);
  trans_gemm_kernel<<<BATCH * (NN / 16) / 8, 256, 0, stream>>>(twb, xs, t_b, yt);
  bn_stats_kernel<<<CH, 256, 0, stream>>>(yt, st);
  finalize_kernel<<<(int)(((size_t)BATCH * CH * NN + 255) / 256), 256, 0,
                    stream>>>(x, yt, st, gamma, beta, out);
}
